// CrossFeatureAffinityPooling_71425306132902
// MI455X (gfx1250) — compile-verified, matrix operand order checked by reading, not run on hardware
//
#include <hip/hip_runtime.h>
#include <hip/hip_bf16.h>
#include <math.h>

#define BB 4
#define CC 256
#define NN 4096          // 64*64
#define GROUPS 32
#define CPG (CC/GROUPS)  // 8
#define EPSV 1e-5f

typedef __attribute__((ext_vector_type(16))) __bf16 v16bf;
typedef __attribute__((ext_vector_type(8)))  float  v8f;
typedef __attribute__((ext_vector_type(4)))  unsigned int u32x4;
typedef unsigned short u16;

union ATile { v16bf v; u16 h[16]; u32x4 q[2]; };

__device__ __forceinline__ u16 f2bf(float f) {
    unsigned u = __float_as_uint(f);
    unsigned r = u + 0x7FFFu + ((u >> 16) & 1u);   // round-to-nearest-even
    return (u16)(r >> 16);
}
__device__ __forceinline__ float bf2f(u16 h) {
    return __uint_as_float(((unsigned)h) << 16);
}

// Load a 16x32 bf16 WMMA A-operand (or 32x16 B-operand, which mirrors it with
// row->col) from row-major [row][k] memory with K contiguous.
// Per ISA 7.12.2: lanes 0-15 hold K={0..7,16..23} of row M=lane,
// lanes 16-31 hold K={8..15,24..31} of row M=lane-16.
__device__ __forceinline__ v16bf load_tile(const u16* __restrict__ base, int stride,
                                           int row0, int k0, int lane) {
    int lr = lane & 15, hf = lane >> 4;
    const u16* p = base + (size_t)(row0 + lr) * stride + (k0 + hf * 8);
    ATile t;
    t.q[0] = *(const u32x4*)(p);
    t.q[1] = *(const u32x4*)(p + 16);
    return t.v;
}

__device__ __forceinline__ v8f wmma_bf(v16bf a, v16bf b, v8f c) {
    return __builtin_amdgcn_wmma_f32_16x16x32_bf16(false, a, false, b, (short)0, c,
                                                   false, false);
}

// ---------------------------------------------------------------- K0: convert
// Read H,U f32 [b][c][n]; emit bf16 H^T,U^T as [b][n][c] (LDS transpose) and
// bf16 U in natural [b][c][n].
__global__ void k0_convert(const float* __restrict__ H, const float* __restrict__ U,
                           u16* __restrict__ HbT, u16* __restrict__ UbT,
                           u16* __restrict__ Ub) {
    __shared__ float lh[32][33];
    __shared__ float lu[32][33];
    int b = blockIdx.z;
    int n0 = blockIdx.x * 32;
    int c0 = blockIdx.y * 32;
    int t = threadIdx.x;
    const size_t bo = (size_t)b * CC * NN;
#pragma unroll
    for (int i = 0; i < 4; ++i) {
        int cl = (t >> 5) + i * 8;
        int nl = t & 31;
        size_t idx = bo + (size_t)(c0 + cl) * NN + n0 + nl;
        float hv = H[idx];
        float uv = U[idx];
        lh[cl][nl] = hv;
        lu[cl][nl] = uv;
        Ub[idx] = f2bf(uv);
    }
    __syncthreads();
#pragma unroll
    for (int i = 0; i < 4; ++i) {
        int nl = (t >> 5) + i * 8;
        int cl = t & 31;
        size_t idx = bo + (size_t)(n0 + nl) * CC + c0 + cl;
        HbT[idx] = f2bf(lh[cl][nl]);
        UbT[idx] = f2bf(lu[cl][nl]);
    }
}

__global__ void k0_weights(const float* __restrict__ WH, const float* __restrict__ WU,
                           u16* __restrict__ WbH, u16* __restrict__ WbK) {
    int i = blockIdx.x * blockDim.x + threadIdx.x;
    if (i < CC * CC) {
        WbH[i] = f2bf(WH[i]);
        WbK[i] = f2bf(WU[i]);
    }
}

// ---------------------------------------------------------------- K1: projection
// Y^T[b][n][o] = sum_c X^T[b][n][c] * W[o][c] + bias[o]   (WMMA: M=n,K=c,N=o)
__global__ void k1_proj(const u16* __restrict__ Xt, const u16* __restrict__ Wb,
                        const float* __restrict__ bias, u16* __restrict__ Yt) {
    int lane = threadIdx.x & 31;
    int wid = threadIdx.x >> 5;
    int gw = blockIdx.x * 4 + wid;       // 0..1023
    int ntile = gw >> 2;                 // 256 n-tiles
    int oblk = (gw & 3) * 64;            // 4 output-channel blocks of 64
    int b = blockIdx.y;
    const size_t bo = (size_t)b * NN * CC;
    const u16* X = Xt + bo;
    int n0 = ntile * 16;
    int lr = lane & 15, hf = lane >> 4;

    v8f acc[4];
#pragma unroll
    for (int ct = 0; ct < 4; ++ct) {
        float bv = bias[oblk + ct * 16 + lr];
        acc[ct] = (v8f){bv, bv, bv, bv, bv, bv, bv, bv};
    }
#pragma unroll
    for (int kc = 0; kc < 8; ++kc) {
        v16bf a = load_tile(X, CC, n0, kc * 32, lane);
#pragma unroll
        for (int ct = 0; ct < 4; ++ct) {
            v16bf w = load_tile(Wb, CC, oblk + ct * 16, kc * 32, lane);
            acc[ct] = wmma_bf(a, w, acc[ct]);
        }
    }
#pragma unroll
    for (int ct = 0; ct < 4; ++ct)
#pragma unroll
        for (int r = 0; r < 8; ++r) {
            int n = n0 + r + 8 * hf;
            int o = oblk + ct * 16 + lr;
            Yt[bo + (size_t)n * CC + o] = f2bf(acc[ct][r]);
        }
}

// ---------------------------------------------------------------- K2a: scores
// S[n][m] = sum_c qT[n][c] * kT[m][c]   (per batch; store bf16)
__global__ void k2a_scores(const u16* __restrict__ qT, const u16* __restrict__ kT,
                           u16* __restrict__ S) {
    int lane = threadIdx.x & 31;
    int wid = threadIdx.x >> 5;
    int gw = blockIdx.x * 4 + wid;       // 0..2047
    int ntile = gw >> 3;                 // 256 n-tiles
    int mchunk = gw & 7;                 // 8 chunks of 32 m-tiles
    int n0 = ntile * 16;
    int lr = lane & 15, hf = lane >> 4;

    v16bf a[8];
#pragma unroll
    for (int kc = 0; kc < 8; ++kc) a[kc] = load_tile(qT, CC, n0, kc * 32, lane);

    for (int mt = mchunk * 32; mt < mchunk * 32 + 32; ++mt) {
        int m0 = mt * 16;
        if (mt + 1 < 256) __builtin_prefetch(kT + (size_t)(m0 + 16) * CC, 0, 0);
        v8f acc = (v8f){0.f, 0.f, 0.f, 0.f, 0.f, 0.f, 0.f, 0.f};
#pragma unroll
        for (int kc = 0; kc < 8; ++kc) {
            v16bf bt = load_tile(kT, CC, m0, kc * 32, lane);
            acc = wmma_bf(a[kc], bt, acc);
        }
#pragma unroll
        for (int r = 0; r < 8; ++r) {
            int n = n0 + r + 8 * hf;
            S[(size_t)n * NN + m0 + lr] = f2bf(acc[r]);
        }
    }
}

// ---------------------------------------------------------------- K2b: row stats
__global__ void k2b_stats(const u16* __restrict__ S, float* __restrict__ rmax,
                          float* __restrict__ rsum) {
    __shared__ float red[256];
    int row = blockIdx.x;
    int t = threadIdx.x;
    const u16* p = S + (size_t)row * NN;
    float v[16];
    float mx = -3.4e38f;
#pragma unroll
    for (int i = 0; i < 16; ++i) {
        v[i] = bf2f(p[t + i * 256]);
        mx = fmaxf(mx, v[i]);
    }
    red[t] = mx;
    __syncthreads();
    for (int s = 128; s > 0; s >>= 1) {
        if (t < s) red[t] = fmaxf(red[t], red[t + s]);
        __syncthreads();
    }
    mx = red[0];
    __syncthreads();
    float sm = 0.f;
#pragma unroll
    for (int i = 0; i < 16; ++i) sm += __expf(v[i] - mx);
    red[t] = sm;
    __syncthreads();
    for (int s = 128; s > 0; s >>= 1) {
        if (t < s) red[t] += red[t + s];
        __syncthreads();
    }
    if (t == 0) {
        rmax[row] = mx;
        rsum[row] = red[0];
    }
}

// ---------------------------------------------------------------- K2c: O = softmax(S) @ U^T
// O[n][c] = sum_m exp(S[n][m]-max_n)/sum_n * Ub[c][m]  (WMMA: M=n,K=m,N=c)
// All 16 A-operand elements in a lane share row n0+(lane&15) -> row stats are lane scalars.
__global__ void k2c_out(const u16* __restrict__ S, const u16* __restrict__ Ub,
                        const float* __restrict__ rmax, const float* __restrict__ rsum,
                        float* __restrict__ O) {
    int lane = threadIdx.x & 31;
    int wid = threadIdx.x >> 5;
    int gw = blockIdx.x * 4 + wid;       // 0..1023
    int ntile = gw >> 2;
    int cblk = (gw & 3) * 64;
    int n0 = ntile * 16;
    int lr = lane & 15, hf = lane >> 4;

    float mx = rmax[n0 + lr];
    float rinv = 1.0f / rsum[n0 + lr];

    v8f acc[4];
#pragma unroll
    for (int ct = 0; ct < 4; ++ct) acc[ct] = (v8f){0.f, 0.f, 0.f, 0.f, 0.f, 0.f, 0.f, 0.f};

    for (int mk = 0; mk < 128; ++mk) {
        ATile t;
        t.v = load_tile(S, NN, n0, mk * 32, lane);
#pragma unroll
        for (int i = 0; i < 16; ++i) {
            float pv = __expf(bf2f(t.h[i]) - mx) * rinv;
            t.h[i] = f2bf(pv);
        }
#pragma unroll
        for (int ct = 0; ct < 4; ++ct) {
            v16bf bt = load_tile(Ub, NN, cblk + ct * 16, mk * 32, lane);
            acc[ct] = wmma_bf(t.v, bt, acc[ct]);
        }
    }
#pragma unroll
    for (int ct = 0; ct < 4; ++ct)
#pragma unroll
        for (int r = 0; r < 8; ++r) {
            int n = n0 + r + 8 * hf;
            O[(size_t)n * CC + cblk + ct * 16 + lr] = acc[ct][r];
        }
}

// ---------------------------------------------------------------- K3: GroupNorm + residual
// O is [b][n][c]; output is [b][c][n] with gn*w+b + Hand.
__global__ void k3_gn(const float* __restrict__ O, const float* __restrict__ Hand,
                      const float* __restrict__ gnw, const float* __restrict__ gnb,
                      float* __restrict__ out) {
    __shared__ float rs[256], rq[256];
    int bg = blockIdx.x;                 // b*GROUPS + g
    int b = bg >> 5, g = bg & 31;
    int t = threadIdx.x;
    const float* Ob = O + (size_t)b * NN * CC;

    float s = 0.f, q = 0.f;
    for (int i = t; i < CPG * NN; i += 256) {
        int cl = i & 7, n = i >> 3;
        float v = Ob[(size_t)n * CC + g * CPG + cl];
        s += v;
        q += v * v;
    }
    rs[t] = s;
    rq[t] = q;
    __syncthreads();
    for (int st = 128; st > 0; st >>= 1) {
        if (t < st) {
            rs[t] += rs[t + st];
            rq[t] += rq[t + st];
        }
        __syncthreads();
    }
    float inv_n = 1.0f / (float)(CPG * NN);
    float mean = rs[0] * inv_n;
    float var = rq[0] * inv_n - mean * mean;
    float rstd = rsqrtf(var + EPSV);
    __syncthreads();

    for (int i = t; i < CPG * NN; i += 256) {
        int cl = i >> 12;                 // channel-in-group (0..7)
        int n = i & (NN - 1);             // n fastest -> coalesced writes
        int c = g * CPG + cl;
        float v = (Ob[(size_t)n * CC + c] - mean) * rstd;
        size_t oi = ((size_t)b * CC + c) * NN + n;
        out[oi] = v * gnw[c] + gnb[c] + Hand[oi];
    }
}

// ---------------------------------------------------------------- launch
extern "C" void kernel_launch(void* const* d_in, const int* in_sizes, int n_in,
                              void* d_out, int out_size, void* d_ws, size_t ws_size,
                              hipStream_t stream) {
    (void)in_sizes; (void)n_in; (void)out_size; (void)ws_size;
    const float* Hand = (const float*)d_in[0];
    const float* U    = (const float*)d_in[1];
    const float* WHw  = (const float*)d_in[2];
    const float* WHb  = (const float*)d_in[3];
    const float* WUw  = (const float*)d_in[4];
    const float* WUb  = (const float*)d_in[5];
    const float* gnw  = (const float*)d_in[6];
    const float* gnb  = (const float*)d_in[7];
    float* out = (float*)d_out;

    char* ws = (char*)d_ws;
    size_t off = 0;
    auto alloc = [&](size_t bytes) {
        void* p = ws + off;
        off += (bytes + 255) & ~(size_t)255;
        return p;
    };
    u16* HbT  = (u16*)alloc((size_t)BB * NN * CC * 2);   // H^T bf16 [b][n][c]
    u16* UbT  = (u16*)alloc((size_t)BB * NN * CC * 2);   // U^T bf16 [b][n][c]
    u16* Ub   = (u16*)alloc((size_t)BB * CC * NN * 2);   // U bf16   [b][c][n]
    u16* WbH  = (u16*)alloc((size_t)CC * CC * 2);
    u16* WbK  = (u16*)alloc((size_t)CC * CC * 2);
    u16* qT   = (u16*)alloc((size_t)BB * NN * CC * 2);   // q^T bf16 [b][n][c]
    u16* kT   = (u16*)alloc((size_t)BB * NN * CC * 2);   // k^T bf16 [b][n][c]
    u16* S    = (u16*)alloc((size_t)NN * NN * 2);        // per-batch scores (reused)
    float* rmax = (float*)alloc((size_t)NN * 4);
    float* rsum = (float*)alloc((size_t)NN * 4);
    float* O  = (float*)alloc((size_t)BB * NN * CC * 4); // out pre-GN [b][n][c]

    k0_convert<<<dim3(NN / 32, CC / 32, BB), 256, 0, stream>>>(Hand, U, HbT, UbT, Ub);
    k0_weights<<<(CC * CC + 255) / 256, 256, 0, stream>>>(WHw, WUw, WbH, WbK);
    k1_proj<<<dim3(256, BB), 128, 0, stream>>>(HbT, WbH, WHb, qT);
    k1_proj<<<dim3(256, BB), 128, 0, stream>>>(UbT, WbK, WUb, kT);
    for (int b = 0; b < BB; ++b) {
        size_t po = (size_t)b * NN * CC;
        k2a_scores<<<512, 128, 0, stream>>>(qT + po, kT + po, S);
        k2b_stats<<<NN, 256, 0, stream>>>(S, rmax, rsum);
        k2c_out<<<256, 128, 0, stream>>>(S, Ub + po, rmax, rsum, (float*)O + po);
    }
    k3_gn<<<BB * GROUPS, 256, 0, stream>>>(O, Hand, gnw, gnb, out);
}
